// SpatialGAT_90795608637704
// MI455X (gfx1250) — matrix operand order repaired and checked by reading, not compiled, not run on hardware
//
#include <hip/hip_runtime.h>
#include <hip/hip_bf16.h>
#include <stdint.h>

// ---------------------------------------------------------------------------
// Problem constants (from reference)
// ---------------------------------------------------------------------------
#define NNODES 20000
#define MPAD   20224               // padded to multiple of 256 (block M tile)
#define NEDGES 320000
#define ETOT   (NEDGES + NNODES)   // edges + self loops = 340000
#define FIN    432
#define KPAD1  448                 // 432 padded to multiple of 32
#define HC1    512                 // 4 heads * 128 ch
#define H1     4
#define C1     128
#define HC2    64                  // 1 head * 64 ch
#define H2     1
#define C2     64

// ---------------------------------------------------------------------------
// Types for WMMA
// ---------------------------------------------------------------------------
typedef __bf16 bf16;
typedef __attribute__((ext_vector_type(16))) __bf16 v16bf;
typedef __attribute__((ext_vector_type(8)))  __bf16 v8bf;
typedef __attribute__((ext_vector_type(8)))  float  v8f;

union ABFrag { v16bf v; v8bf h[2]; short s[16]; };

__device__ __forceinline__ bf16 f2bf(float f) {
  unsigned u = __float_as_uint(f);
  unsigned r = (u + 0x7FFFu + ((u >> 16) & 1u)) >> 16;  // RNE to bf16
  unsigned short hs = (unsigned short)r;
  bf16 out;
  __builtin_memcpy(&out, &hs, 2);
  return out;
}

__device__ __forceinline__ float leaky(float v) { return v > 0.f ? v : 0.2f * v; }

__device__ __forceinline__ void atomicMaxF(float* addr, float v) {
  // classic signed/unsigned split: positives via int max, negatives via uint min
  if (v >= 0.f) atomicMax((int*)addr, __float_as_int(v));
  else          atomicMin((unsigned int*)addr, __float_as_uint(v));
}

// ---------------------------------------------------------------------------
// Workspace layout (bytes). H1BF aliases XW1 (xw1 dead before h1bf written).
// A-side (bf16) and C-side (f32) GEMM buffers padded to MPAD rows.
// ---------------------------------------------------------------------------
enum : size_t {
  OFF_XBF   = 0,            // 20224*448 bf16   = 18,120,704
  OFF_W1SW  = 18120704,     // 448*512 bf16     =    458,752
  OFF_XW1   = 18579456,     // 20224*512 f32    = 41,418,752 (reused: H1BF bf16 padded)
  OFF_AS1   = 59998208,     // 20000*4 f32
  OFF_AD1   = 60318208,
  OFF_M1    = 60638208,
  OFF_DEN1  = 60958208,
  OFF_EEXP1 = 61278208,     // 340000*4 f32     = 5,440,000
  OFF_OUT1  = 66718208,     // 20000*512 f32    = 40,960,000
  OFF_W2SW  = 107678208,    // 512*64 bf16
  OFF_XW2   = 107743744,    // 20224*64 f32     = 5,177,344
  OFF_AS2   = 112921088,
  OFF_AD2   = 113001088,
  OFF_M2    = 113081088,
  OFF_DEN2  = 113161088,
  OFF_EEXP2 = 113241088,    // 340000 f32
  OFF_OUT2  = 114601088,    // 20000*64 f32
  WS_TOTAL  = 119721088
};

// ---------------------------------------------------------------------------
// Prep kernels
// ---------------------------------------------------------------------------
__global__ __launch_bounds__(256) void k_fill(float* p, float v, int n) {
  int i = blockIdx.x * 256 + threadIdx.x;
  if (i < n) p[i] = v;
}

// x [NNODES, FIN] f32 -> bf16 padded [MPAD, KPAD1] (pad rows & cols zeroed)
__global__ __launch_bounds__(256) void k_cvt_pad(const float* __restrict__ x,
                                                 bf16* __restrict__ dst) {
  int i = blockIdx.x * 256 + threadIdx.x;
  if (i >= MPAD * KPAD1) return;
  int row = i / KPAD1, col = i - row * KPAD1;
  float v = (row < NNODES && col < FIN) ? x[row * FIN + col] : 0.f;
  dst[i] = f2bf(v);
}

// W [Korig, N] f32 row-major -> bf16 in WMMA B-fragment order:
// out[((kb*nTiles + nt)*32 + lane)*16 + j]  where
//   K = kb*32 + (j<8 ? (lane>>4)*8 + j : 16 + (lane>>4)*8 + j-8),  n = nt*16 + (lane&15)
__global__ __launch_bounds__(256) void k_swizzle_w(const float* __restrict__ W,
                                                   bf16* __restrict__ out,
                                                   int Korig, int N, int total) {
  int i = blockIdx.x * 256 + threadIdx.x;
  if (i >= total) return;
  int j    = i & 15;
  int lane = (i >> 4) & 31;
  int tile = i >> 9;
  int nTiles = N >> 4;
  int nt = tile % nTiles;
  int kb = tile / nTiles;
  int K = kb * 32 + (j < 8 ? (lane >> 4) * 8 + j : 16 + (lane >> 4) * 8 + (j - 8));
  int n = nt * 16 + (lane & 15);
  float v = (K < Korig) ? W[(size_t)K * N + n] : 0.f;
  out[i] = f2bf(v);
}

// ---------------------------------------------------------------------------
// WMMA bf16 GEMM: C[MPAD,N] = A[MPAD,Kt*32] * Bsw (pre-swizzled).
// All M dims pre-padded -> branch-free hot loop.
// block = 256 threads = 8 waves; each wave computes a 32x64 C tile
// (2 A-fragments x 4 B-fragments -> 8 WMMAs per K-step).
// ---------------------------------------------------------------------------
__global__ __launch_bounds__(256)
void gat_gemm_bf16(const bf16* __restrict__ A, const bf16* __restrict__ Bsw,
                   float* __restrict__ C, int Kt, int N, int lda) {
  const int lane = threadIdx.x & 31;
  const int wave = threadIdx.x >> 5;
  const int r    = lane & 15;
  const int half = lane >> 4;
  const int nTiles  = N >> 4;
  const int rowBase = blockIdx.x * 256 + wave * 32;
  const int ntBase  = blockIdx.y * 4;

  v8f acc[8];
#pragma unroll
  for (int t = 0; t < 8; ++t) acc[t] = v8f{};

  const bf16* aRow0 = A + (size_t)(rowBase + r) * lda;
  const bf16* aRow1 = A + (size_t)(rowBase + 16 + r) * lda;

  for (int kb = 0; kb < Kt; ++kb) {
    // 16-bit A 16x32 layout: elems 0..7  -> K = kb*32 + half*8 + j
    //                        elems 8..15 -> K = kb*32 + 16 + half*8 + j
    ABFrag a0, a1;
    a0.h[0] = *(const v8bf*)(aRow0 + kb * 32 + half * 8);
    a0.h[1] = *(const v8bf*)(aRow0 + kb * 32 + 16 + half * 8);
    a1.h[0] = *(const v8bf*)(aRow1 + kb * 32 + half * 8);
    a1.h[1] = *(const v8bf*)(aRow1 + kb * 32 + 16 + half * 8);

    const bf16* bBase = Bsw + (((size_t)kb * nTiles + ntBase) * 32 + lane) * 16;
#pragma unroll
    for (int t = 0; t < 4; ++t) {
      ABFrag b;
      b.h[0] = *(const v8bf*)(bBase + (size_t)t * 512);
      b.h[1] = *(const v8bf*)(bBase + (size_t)t * 512 + 8);
      acc[t] = __builtin_amdgcn_wmma_f32_16x16x32_bf16(
          false, a0.v, false, b.v, (short)0, acc[t], false, false);
      acc[4 + t] = __builtin_amdgcn_wmma_f32_16x16x32_bf16(
          false, a1.v, false, b.v, (short)0, acc[4 + t], false, false);
    }
  }

  // C layout: elem i of v8f -> row = rowGroupBase + half*8 + i, col = tileCol + r
#pragma unroll
  for (int g = 0; g < 2; ++g) {
    const int rOut = rowBase + g * 16 + half * 8;
#pragma unroll
    for (int t = 0; t < 4; ++t) {
      const int col = (ntBase + t) * 16 + r;
#pragma unroll
      for (int i = 0; i < 8; ++i) {
        C[(size_t)(rOut + i) * N + col] = acc[g * 4 + t][i];
      }
    }
  }
}

// ---------------------------------------------------------------------------
// Attention logits per node/head: alpha_s/alpha_d = sum_c xW * a
// ---------------------------------------------------------------------------
__global__ __launch_bounds__(256)
void k_alpha(const float* __restrict__ xW, const float* __restrict__ a_s,
             const float* __restrict__ a_d, float* __restrict__ outS,
             float* __restrict__ outD, int nh, int H, int C) {
  int i = blockIdx.x * 256 + threadIdx.x;   // i = node*H + h
  if (i >= nh) return;
  int h = i % H;
  const float* xr = xW + (size_t)i * C;
  const float* as = a_s + (size_t)h * C;
  const float* ad = a_d + (size_t)h * C;
  float ss = 0.f, sd = 0.f;
  for (int c = 0; c < C; ++c) { float v = xr[c]; ss += v * as[c]; sd += v * ad[c]; }
  outS[i] = ss; outD[i] = sd;
}

// ---------------------------------------------------------------------------
// Edge passes (segment softmax over incoming edges of dst)
// ---------------------------------------------------------------------------
__device__ __forceinline__ void edge_sd(const int* __restrict__ ei, int e, int& s, int& d) {
  if (e < NEDGES) { s = ei[e]; d = ei[NEDGES + e]; }
  else            { s = e - NEDGES; d = s; }        // self loops appended
}

__global__ __launch_bounds__(256)
void k_edge_max(const int* __restrict__ ei, const float* __restrict__ as,
                const float* __restrict__ ad, float* __restrict__ m, int H) {
  int e = blockIdx.x * 256 + threadIdx.x;
  if (e >= ETOT) return;
  int s, d; edge_sd(ei, e, s, d);
  for (int h = 0; h < H; ++h) {
    float v = leaky(as[s * H + h] + ad[d * H + h]);
    atomicMaxF(&m[d * H + h], v);
  }
}

__global__ __launch_bounds__(256)
void k_edge_expsum(const int* __restrict__ ei, const float* __restrict__ as,
                   const float* __restrict__ ad, const float* __restrict__ m,
                   float* __restrict__ eexp, float* __restrict__ den, int H) {
  int e = blockIdx.x * 256 + threadIdx.x;
  if (e >= ETOT) return;
  int s, d; edge_sd(ei, e, s, d);
  for (int h = 0; h < H; ++h) {
    float v = leaky(as[s * H + h] + ad[d * H + h]);
    float w = __expf(v - m[d * H + h]);
    eexp[(size_t)e * H + h] = w;
    atomicAdd(&den[d * H + h], w);
  }
}

// out[dst] += xW[src] * alpha ; one thread handles 4 channels (float4 gather)
__global__ __launch_bounds__(256)
void k_edge_agg(const int* __restrict__ ei, const float* __restrict__ xW,
                const float* __restrict__ eexp, const float* __restrict__ den,
                float* __restrict__ out, int H, int C) {
  const int HC = H * C;
  const int chunks = HC >> 2;
  long long tid = (long long)blockIdx.x * 256 + threadIdx.x;
  if (tid >= (long long)ETOT * chunks) return;
  int e  = (int)(tid / chunks);
  int c4 = (int)(tid - (long long)e * chunks) * 4;
  int h  = c4 / C;
  int s, d; edge_sd(ei, e, s, d);
  float alpha = eexp[(size_t)e * H + h] / den[d * H + h];
  const float4 xv = *(const float4*)(xW + (size_t)s * HC + c4);
  float* o = out + (size_t)d * HC + c4;
  atomicAdd(o + 0, xv.x * alpha);
  atomicAdd(o + 1, xv.y * alpha);
  atomicAdd(o + 2, xv.z * alpha);
  atomicAdd(o + 3, xv.w * alpha);
}

// ELU(in + b) -> bf16 (feeds next GEMM)
__global__ __launch_bounds__(256)
void k_elu_cvt(const float* __restrict__ in, const float* __restrict__ b,
               bf16* __restrict__ out, int n, int hc) {
  int i = blockIdx.x * 256 + threadIdx.x;
  if (i >= n) return;
  float v = in[i] + b[i % hc];
  v = v > 0.f ? v : (__expf(v) - 1.f);
  out[i] = f2bf(v);
}

// ---------------------------------------------------------------------------
// Final: h2 = elu(out2 + b2); h3 = relu(h2@W3 + b3); y = h3@W4 + b4
// ---------------------------------------------------------------------------
__global__ __launch_bounds__(256)
void k_head(const float* __restrict__ out2, const float* __restrict__ b2,
            const float* __restrict__ W3, const float* __restrict__ b3,
            const float* __restrict__ W4, const float* __restrict__ b4,
            float* __restrict__ y) {
  __shared__ float sW3[64 * 32];
  __shared__ float sW4[32 * 2];
  __shared__ float sb2[64];
  __shared__ float sb3[32];
  __shared__ float sb4[2];
  for (int i = threadIdx.x; i < 64 * 32; i += 256) sW3[i] = W3[i];
  for (int i = threadIdx.x; i < 64;      i += 256) { sW4[i] = W4[i]; sb2[i] = b2[i]; }
  for (int i = threadIdx.x; i < 32;      i += 256) sb3[i] = b3[i];
  if (threadIdx.x < 2) sb4[threadIdx.x] = b4[threadIdx.x];
  __syncthreads();

  int node = blockIdx.x * 256 + threadIdx.x;
  if (node >= NNODES) return;
  float row[64];
  const float* r = out2 + (size_t)node * 64;
#pragma unroll
  for (int c = 0; c < 64; ++c) {
    float v = r[c] + sb2[c];
    row[c] = v > 0.f ? v : (__expf(v) - 1.f);   // elu
  }
  float o0 = sb4[0], o1 = sb4[1];
#pragma unroll
  for (int j = 0; j < 32; ++j) {
    float s = sb3[j];
#pragma unroll
    for (int c = 0; c < 64; ++c) s += row[c] * sW3[c * 32 + j];
    s = s > 0.f ? s : 0.f;                       // relu
    o0 += s * sW4[j * 2 + 0];
    o1 += s * sW4[j * 2 + 1];
  }
  y[(size_t)node * 2 + 0] = o0;
  y[(size_t)node * 2 + 1] = o1;
}

// ---------------------------------------------------------------------------
// Launch
// ---------------------------------------------------------------------------
extern "C" void kernel_launch(void* const* d_in, const int* in_sizes, int n_in,
                              void* d_out, int out_size, void* d_ws, size_t ws_size,
                              hipStream_t stream) {
  const float* x     = (const float*)d_in[0];
  const int*   ei    = (const int*)  d_in[1];
  const float* W1    = (const float*)d_in[2];
  const float* asrc1 = (const float*)d_in[3];
  const float* adst1 = (const float*)d_in[4];
  const float* b1    = (const float*)d_in[5];
  const float* W2    = (const float*)d_in[6];
  const float* asrc2 = (const float*)d_in[7];
  const float* adst2 = (const float*)d_in[8];
  const float* b2    = (const float*)d_in[9];
  const float* W3    = (const float*)d_in[10];
  const float* b3    = (const float*)d_in[11];
  const float* W4    = (const float*)d_in[12];
  const float* b4    = (const float*)d_in[13];

  char* ws = (char*)d_ws;
  bf16*  xbf   = (bf16*) (ws + OFF_XBF);
  bf16*  w1sw  = (bf16*) (ws + OFF_W1SW);
  float* xw1   = (float*)(ws + OFF_XW1);
  bf16*  h1bf  = (bf16*) (ws + OFF_XW1);   // alias: xw1 dead before h1bf written
  float* as1   = (float*)(ws + OFF_AS1);
  float* ad1   = (float*)(ws + OFF_AD1);
  float* m1    = (float*)(ws + OFF_M1);
  float* den1  = (float*)(ws + OFF_DEN1);
  float* eexp1 = (float*)(ws + OFF_EEXP1);
  float* out1  = (float*)(ws + OFF_OUT1);
  bf16*  w2sw  = (bf16*) (ws + OFF_W2SW);
  float* xw2   = (float*)(ws + OFF_XW2);
  float* as2   = (float*)(ws + OFF_AS2);
  float* ad2   = (float*)(ws + OFF_AD2);
  float* m2    = (float*)(ws + OFF_M2);
  float* den2  = (float*)(ws + OFF_DEN2);
  float* eexp2 = (float*)(ws + OFF_EEXP2);
  float* out2  = (float*)(ws + OFF_OUT2);
  float* y     = (float*)d_out;

  dim3 blk(256);
  const float NEG_INF = -__builtin_inff();
#define GRID1(n) dim3(((unsigned)(((long long)(n) + 255) / 256)))

  // ---- Layer 1 -------------------------------------------------------------
  k_cvt_pad  <<<GRID1(MPAD * KPAD1), blk, 0, stream>>>(x, xbf);
  k_swizzle_w<<<GRID1(KPAD1 * HC1),  blk, 0, stream>>>(W1, w1sw, FIN, HC1, KPAD1 * HC1);

  dim3 g1(MPAD / 256, HC1 / 64);
  gat_gemm_bf16<<<g1, blk, 0, stream>>>(xbf, w1sw, xw1, KPAD1 / 32, HC1, KPAD1);

  k_alpha<<<GRID1(NNODES * H1), blk, 0, stream>>>(xw1, asrc1, adst1, as1, ad1,
                                                  NNODES * H1, H1, C1);
  k_fill<<<GRID1(NNODES * H1),  blk, 0, stream>>>(m1,   NEG_INF, NNODES * H1);
  k_fill<<<GRID1(NNODES * H1),  blk, 0, stream>>>(den1, 0.f,     NNODES * H1);
  k_fill<<<GRID1(NNODES * HC1), blk, 0, stream>>>(out1, 0.f,     NNODES * HC1);

  k_edge_max   <<<GRID1(ETOT), blk, 0, stream>>>(ei, as1, ad1, m1, H1);
  k_edge_expsum<<<GRID1(ETOT), blk, 0, stream>>>(ei, as1, ad1, m1, eexp1, den1, H1);
  k_edge_agg   <<<GRID1((long long)ETOT * (HC1 / 4)), blk, 0, stream>>>(
      ei, xw1, eexp1, den1, out1, H1, C1);

  // ELU + bf16 for GEMM2 (writes over dead xw1 region); zero h1bf pad rows
  k_elu_cvt<<<GRID1(NNODES * HC1), blk, 0, stream>>>(out1, b1, h1bf, NNODES * HC1, HC1);
  k_fill<<<GRID1((MPAD - NNODES) * HC1 / 2), blk, 0, stream>>>(
      (float*)(ws + OFF_XW1 + (size_t)NNODES * HC1 * 2), 0.f, (MPAD - NNODES) * HC1 / 2);

  // ---- Layer 2 -------------------------------------------------------------
  k_swizzle_w<<<GRID1(512 * HC2), blk, 0, stream>>>(W2, w2sw, 512, HC2, 512 * HC2);
  dim3 g2(MPAD / 256, HC2 / 64);
  gat_gemm_bf16<<<g2, blk, 0, stream>>>(h1bf, w2sw, xw2, 512 / 32, HC2, 512);

  k_alpha<<<GRID1(NNODES * H2), blk, 0, stream>>>(xw2, asrc2, adst2, as2, ad2,
                                                  NNODES * H2, H2, C2);
  k_fill<<<GRID1(NNODES * H2),  blk, 0, stream>>>(m2,   NEG_INF, NNODES * H2);
  k_fill<<<GRID1(NNODES * H2),  blk, 0, stream>>>(den2, 0.f,     NNODES * H2);
  k_fill<<<GRID1(NNODES * HC2), blk, 0, stream>>>(out2, 0.f,     NNODES * HC2);

  k_edge_max   <<<GRID1(ETOT), blk, 0, stream>>>(ei, as2, ad2, m2, H2);
  k_edge_expsum<<<GRID1(ETOT), blk, 0, stream>>>(ei, as2, ad2, m2, eexp2, den2, H2);
  k_edge_agg   <<<GRID1((long long)ETOT * (HC2 / 4)), blk, 0, stream>>>(
      ei, xw2, eexp2, den2, out2, H2, C2);

  // ---- Head (elu + MLP) ----------------------------------------------------
  k_head<<<GRID1(NNODES), blk, 0, stream>>>(out2, b2, W3, b3, W4, b4, y);
#undef GRID1
}
